// YoloV2_LOSS_37847251813042
// MI455X (gfx1250) — compile-verified, matrix operand order checked by reading, not run on hardware
//
#include <hip/hip_runtime.h>
#include <hip/hip_bf16.h>
#include <math.h>

typedef float     v2f  __attribute__((ext_vector_type(2)));
typedef float     v8f  __attribute__((ext_vector_type(8)));
typedef _Float16  v16h __attribute__((ext_vector_type(16)));

#define NBATCH 512
#define OUTW   13
#define HWC    (13*13)      // 169
#define NA     5
#define NC     80
#define NGT    20
#define NELEM  (HWC*NA)     // 845

// ---------------------------------------------------------------------------
// Kernel 1: one block per sample. Dense LDS target table + masked MSE sums.
// ---------------------------------------------------------------------------
__global__ __launch_bounds__(256) void yolo_loss_main(
    const float* __restrict__ bbox_pred,   // (B, 845, 4)
    const float* __restrict__ iou_pred,    // (B, 845)
    const float* __restrict__ prob_pred,   // (B, 845, 80)
    const float* __restrict__ bbox_gt,     // (B, 20, 4)
    const int*   __restrict__ cls_gt,      // (B, 20)
    const float* __restrict__ anchors,     // (5, 2)
    float*       __restrict__ partials)    // (B,)
{
    __shared__ float4 s_tgtv[NELEM];       // target (tx,ty,sqrt(w),sqrt(h)) per slot
    __shared__ int    s_info[NELEM];       // (cls << 1) | matched
    __shared__ int    s_slot[NGT];
    __shared__ float  s_anch[2 * NA];
    __shared__ float  s_red[3][256];

    const int b   = blockIdx.x;
    const int tid = threadIdx.x;

    if (tid < 2 * NA) s_anch[tid] = anchors[tid];

    // ---- zero-init dense target table ----
    for (int e = tid; e < NELEM; e += 256) {
        s_tgtv[e] = make_float4(0.f, 0.f, 0.f, 0.f);
        s_info[e] = 0;
    }

    // ---- Phase 1: per-GT target construction (threads 0..19) ----
    int   my_slot = -1, my_cls = 0;
    float t0 = 0.f, t1 = 0.f, t2 = 0.f, t3 = 0.f;
    if (tid < NGT) {
        const float* g = bbox_gt + ((size_t)b * NGT + tid) * 4;
        const float gx = g[0], gy = g[1], gw = g[2], gh = g[3];
        const float rg = gw / gh;
        int best = 0;
        float bd = fabsf(anchors[0] / anchors[1] - rg);
        #pragma unroll
        for (int a = 1; a < NA; ++a) {
            const float d = fabsf(anchors[2 * a] / anchors[2 * a + 1] - rg);
            if (d < bd) { bd = d; best = a; }   // strict <: first occurrence wins
        }
        const float cx = gx * (float)OUTW, cy = gy * (float)OUTW;
        const float fx = floorf(cx), fy = floorf(cy);
        const int cell = (int)(fy * (float)OUTW + fx);
        my_slot = cell * NA + best;
        t0 = cx - fx; t1 = cy - fy; t2 = sqrtf(gw); t3 = sqrtf(gh);
        my_cls = cls_gt[(size_t)b * NGT + tid];
        s_slot[tid] = my_slot;
    }
    __syncthreads();

    // ---- Phase 1b: dedup (last GT writing a slot wins), then scatter ----
    if (tid < NGT) {
        int valid = 1;
        for (int tt = tid + 1; tt < NGT; ++tt)
            if (s_slot[tt] == my_slot) valid = 0;
        if (valid) {
            s_tgtv[my_slot] = make_float4(t0, t1, t2, t3);
            s_info[my_slot] = (my_cls << 1) | 1;
        }
    }
    __syncthreads();

    // ---- Phase 2: strided sweep over 845 (cell,anchor) elements ----
    float sb = 0.f, si = 0.f, sc = 0.f;
    for (int e = tid; e < NELEM; e += 256) {
        const int a = e % NA;
        const float4 bpv = *(const float4*)(bbox_pred + ((size_t)b * NELEM + e) * 4);
        const float aw = s_anch[2 * a], ah = s_anch[2 * a + 1];
        const float px = bpv.x, py = bpv.y;
        const float pz = sqrtf(bpv.z * aw), pw = sqrtf(bpv.w * ah);

        // single dense-table lookup (ds_load_b128 + ds_load_b32)
        const float4 t    = s_tgtv[e];
        const int    info = s_info[e];
        const int    matched = info & 1;
        const int    mcls    = info >> 1;

        // bbox term (mask is 0/1; zero when unmatched)
        {
            const float dx = px - t.x, dy = py - t.y, dz = pz - t.z, dw = pw - t.w;
            const float v  = dx * dx + dy * dy + dz * dz + dw * dw;
            sb += matched ? v : 0.f;
        }

        // iou term: iou(bp, built) — built is zeros when unmatched -> iou 0
        {
            const float a1 = pz * pw;
            const float a2 = t.z * t.w;
            float iw = fminf(px + pz, t.x + t.z) - fmaxf(px, t.x);
            float ih = fminf(py + pw, t.y + t.w) - fmaxf(py, t.y);
            iw = fmaxf(iw, 0.f); ih = fmaxf(ih, 0.f);
            const float inter = iw * ih;
            const float iou_b = inter / (a1 + a2 - inter);   // a1 > 0 always
            const float ip    = iou_pred[(size_t)b * NELEM + e];
            const float imask = matched ? 1.f : 0.5f;
            const float d     = imask * (ip - iou_b);
            si += d * d;
        }

        // cls term: only touch prob row when cls_mask == 1 (bandwidth saver)
        if (matched) {
            const float4* pr = (const float4*)(prob_pred + ((size_t)b * NELEM + e) * NC);
            #pragma unroll
            for (int c4 = 0; c4 < NC / 4; ++c4) {
                const float4 p = pr[c4];
                const int cb = c4 * 4;
                const float u0 = (cb + 0 == mcls) ? 1.f : 0.f;
                const float u1 = (cb + 1 == mcls) ? 1.f : 0.f;
                const float u2 = (cb + 2 == mcls) ? 1.f : 0.f;
                const float u3 = (cb + 3 == mcls) ? 1.f : 0.f;
                const float d0 = p.x - u0, d1 = p.y - u1, d2 = p.z - u2, d3 = p.w - u3;
                sc += d0 * d0 + d1 * d1 + d2 * d2 + d3 * d3;
            }
        }
    }

    // ---- Phase 3: block tree reduction ----
    s_red[0][tid] = sb; s_red[1][tid] = si; s_red[2][tid] = sc;
    __syncthreads();
    for (int s = 128; s > 0; s >>= 1) {
        if (tid < s) {
            s_red[0][tid] += s_red[0][tid + s];
            s_red[1][tid] += s_red[1][tid + s];
            s_red[2][tid] += s_red[2][tid + s];
        }
        __syncthreads();
    }
    if (tid == 0) {
        const float inv_b = 1.f / ((float)NBATCH * (float)NELEM * 4.f);
        const float inv_i = 1.f / ((float)NBATCH * (float)NELEM);
        const float inv_c = 1.f / ((float)NBATCH * (float)NELEM * (float)NC);
        partials[b] = 5.f * s_red[0][0] * inv_b + s_red[1][0] * inv_i + s_red[2][0] * inv_c;
    }
}

// ---------------------------------------------------------------------------
// Kernel 2: one wave32 reduces 512 partials via WMMA (A x ones = row sums).
// EXEC is all ones (32 threads, no divergence before the WMMA).
// ---------------------------------------------------------------------------
__global__ __launch_bounds__(32) void yolo_loss_reduce(
    const float* __restrict__ partials, float* __restrict__ out)
{
    const int l = threadIdx.x;
    float s;

#if __has_builtin(__builtin_amdgcn_wmma_f32_16x16x4_f32)
    // f32 path: 8 chunks of 64 values through V_WMMA_F32_16X16X4_F32
    v8f c = {};
    v2f ones; ones.x = 1.f; ones.y = 1.f;
    #pragma unroll
    for (int k = 0; k < 8; ++k) {
        const float2 p = *(const float2*)(partials + l * 16 + 2 * k);
        v2f av; av.x = p.x; av.y = p.y;
        c = __builtin_amdgcn_wmma_f32_16x16x4_f32(
                /*neg_a=*/false, av, /*neg_b=*/false, ones,
                /*c_mod=*/(short)0, c, /*reuse_a=*/false, /*reuse_b=*/false);
    }
    s = c[0] + c[1] + c[2] + c[3] + c[4] + c[5] + c[6] + c[7];
#else
    // fallback: all 512 values in one V_WMMA_F32_16X16X32_F16 (A is 16x32 f16)
    v16h av, ones;
    #pragma unroll
    for (int k = 0; k < 16; ++k) {
        av[k]   = (_Float16)partials[l * 16 + k];
        ones[k] = (_Float16)1.0f;
    }
    v8f c = {};
    c = __builtin_amdgcn_wmma_f32_16x16x32_f16(
            false, av, false, ones, (short)0, c, false, false);
    s = c[0] + c[1] + c[2] + c[3] + c[4] + c[5] + c[6] + c[7];
#endif

    // lanes 0-15 hold sum over M=0..7, lanes 16-31 hold sum over M=8..15
    s += __shfl_xor(s, 16);
    if (l == 0) out[0] = s;
}

// ---------------------------------------------------------------------------
extern "C" void kernel_launch(void* const* d_in, const int* in_sizes, int n_in,
                              void* d_out, int out_size, void* d_ws, size_t ws_size,
                              hipStream_t stream) {
    const float* bbox_pred = (const float*)d_in[0];
    const float* iou_pred  = (const float*)d_in[1];
    const float* prob_pred = (const float*)d_in[2];
    const float* bbox_gt   = (const float*)d_in[3];
    const int*   cls_gt    = (const int*)d_in[4];
    const float* anchors   = (const float*)d_in[5];
    float*       partials  = (float*)d_ws;   // 512 floats

    yolo_loss_main<<<NBATCH, 256, 0, stream>>>(
        bbox_pred, iou_pred, prob_pred, bbox_gt, cls_gt, anchors, partials);
    yolo_loss_reduce<<<1, 32, 0, stream>>>(partials, (float*)d_out);
}
